// Model_EdgeCycle_batch_39745627357900
// MI455X (gfx1250) — compile-verified, matrix operand order checked by reading, not run on hardware
//
#include <hip/hip_runtime.h>
#include <hip/hip_bf16.h>
#include <stdint.h>

typedef __bf16 bf16_t;
typedef __attribute__((ext_vector_type(16))) __bf16 v16bf;
typedef __attribute__((ext_vector_type(8)))  __bf16 v8bf;
typedef __attribute__((ext_vector_type(8)))  float   v8f;

namespace {
constexpr int kH    = 64;
constexpr int kN    = 40000;
constexpr int kE    = 50000;
constexpr int kE2   = 100000;
constexpr int kC5   = 10000;
constexpr int kR5   = 50000;   // 5*C5
constexpr int kR6   = 60000;   // 6*C6
constexpr int kR    = 110000;  // R5+R6
constexpr int kB    = 256;
constexpr int kOut  = 11;
}

enum { IM_IDENT = 0, IM_ARR = 1, IM_HALF = 2, IM_FIFTH = 3, IM_SIXTH = 4, IM_CYC = 5 };

__device__ __forceinline__ int map_row(int i, int mode, const int* __restrict__ arr) {
  switch (mode) {
    case IM_IDENT: return i;
    case IM_ARR:   return arr[i];
    case IM_HALF:  return i >> 1;
    case IM_FIFTH: return i / 5;
    case IM_SIXTH: return i / 6;
    default:       return (i < kR5) ? (i / 5) : (kC5 + (i - kR5) / 6);   // cyc domains
  }
}

// ---------------- irregular data movement ----------------

// dst[dstrow(i)][dst_col + c] += src[srcrow(i)][src_col + c]
__global__ void k_gscatter_add(float* __restrict__ dst, int dst_ld, int dst_col,
                               const int* __restrict__ dst_arr, int dst_mode,
                               const float* __restrict__ src, int src_ld, int src_col,
                               const int* __restrict__ src_arr, int src_mode,
                               int rows, int C) {
  long long t = (long long)blockIdx.x * blockDim.x + threadIdx.x;
  long long total = (long long)rows * C;
  if (t >= total) return;
  int i = (int)(t / C), c = (int)(t % C);
  int dr = map_row(i, dst_mode, dst_arr);
  int sr = map_row(i, src_mode, src_arr);
  atomicAdd(&dst[(long long)dr * dst_ld + dst_col + c],
            src[(long long)sr * src_ld + src_col + c]);
}

// dst[i][dst_col + c] = src[row(i)][c]   (also plain copy with IM_IDENT)
__global__ void k_gather(float* __restrict__ dst, int dst_ld, int dst_col,
                         const float* __restrict__ src, int src_ld,
                         const int* __restrict__ idx, int idx_mode,
                         int rows, int C) {
  long long t = (long long)blockIdx.x * blockDim.x + threadIdx.x;
  long long total = (long long)rows * C;
  if (t >= total) return;
  int i = (int)(t / C), c = (int)(t % C);
  int sr = map_row(i, idx_mode, idx);
  dst[(long long)i * dst_ld + dst_col + c] = src[(long long)sr * src_ld + c];
}

__global__ void k_f32_to_bf16(bf16_t* __restrict__ dst, const float* __restrict__ src,
                              long long n) {
  long long t = (long long)blockIdx.x * blockDim.x + threadIdx.x;
  if (t < n) dst[t] = (bf16_t)src[t];
}

// W [K,Nact] fp32 row-major -> Wt [Npad,K] bf16 (transposed, zero padded)
__global__ void k_wprep(bf16_t* __restrict__ Wt, const float* __restrict__ W,
                        int K, int Nact, int Npad) {
  long long t = (long long)blockIdx.x * blockDim.x + threadIdx.x;
  long long total = (long long)Npad * K;
  if (t >= total) return;
  int n = (int)(t / K), k = (int)(t % K);
  Wt[t] = (n < Nact) ? (bf16_t)W[(long long)k * Nact + n] : (bf16_t)0.0f;
}

// ---------------- bf16 WMMA GEMM ----------------
// C[M,Nact] (ld=ldc) = A[M,K](bf16) x Wt[Npad,K](bf16, N-major), f32 accum.
// B tile (64 cols x K) staged in LDS once per workgroup (row pad +8 elems for
// bank-conflict-free fragment reads). A double-buffered in registers.
// mode: 0 = plain store, 1 = +bias then ReLU, 2 = +bias
__global__ __launch_bounds__(256)
void k_gemm_bf16(float* __restrict__ Cmat, int ldc,
                 const bf16_t* __restrict__ A, const bf16_t* __restrict__ Wt,
                 int M, int Nact, int K,
                 const float* __restrict__ bias, int mode) {
  extern __shared__ bf16_t Blds[];
  const int ldsK = K + 8;

  const int wave = threadIdx.x >> 5;
  const int lane = threadIdx.x & 31;
  const int half = lane >> 4;      // ISA: lanes16-31 hold K offset +8
  const int l15  = lane & 15;
  const int tileM = blockIdx.x * 128;
  const int tileN = blockIdx.y * 64;

  // ---- stage B tile (64 rows of Wt, each K bf16) into LDS, padded rows ----
  {
    const bf16_t* Bsrc = Wt + (long long)tileN * K;
    const int vecsPerRow = K >> 3;             // K multiple of 32
    for (int idx = threadIdx.x; idx < 64 * vecsPerRow; idx += 256) {
      int n = idx / vecsPerRow;
      int v = idx - n * vecsPerRow;
      *(v8bf*)(Blds + n * ldsK + v * 8) =
          *(const v8bf*)(Bsrc + (long long)n * K + v * 8);
    }
  }
  __syncthreads();

  int rowA = tileM + wave * 16 + l15;
  if (rowA >= M) rowA = M - 1;                 // clamp; stores are masked
  const bf16_t* Abase = A + (long long)rowA * K + 8 * half;

  const bf16_t* Bl[4];
#pragma unroll
  for (int s = 0; s < 4; ++s)
    Bl[s] = Blds + (s * 16 + l15) * ldsK + 8 * half;

  v8f acc[4] = {v8f{0}, v8f{0}, v8f{0}, v8f{0}};

  // ---- K loop: A register double-buffer, B fragments from LDS ----
  v8bf alo = *(const v8bf*)(Abase);
  v8bf ahi = *(const v8bf*)(Abase + 16);
  for (int k = 0; k < K; k += 32) {
    int kn = (k + 32 < K) ? (k + 32) : 0;      // wrap: harmless extra load
    v8bf nlo = *(const v8bf*)(Abase + kn);
    v8bf nhi = *(const v8bf*)(Abase + kn + 16);

    v16bf a = __builtin_shufflevector(alo, ahi,
        0,1,2,3,4,5,6,7,8,9,10,11,12,13,14,15);

    v16bf b[4];
#pragma unroll
    for (int s = 0; s < 4; ++s) {
      v8bf blo = *(const v8bf*)(Bl[s] + k);
      v8bf bhi = *(const v8bf*)(Bl[s] + k + 16);
      b[s] = __builtin_shufflevector(blo, bhi,
          0,1,2,3,4,5,6,7,8,9,10,11,12,13,14,15);
    }
#pragma unroll
    for (int s = 0; s < 4; ++s)
      acc[s] = __builtin_amdgcn_wmma_f32_16x16x32_bf16(
          false, a, false, b[s], (short)0, acc[s], false, false);

    alo = nlo; ahi = nhi;
  }

#pragma unroll
  for (int s = 0; s < 4; ++s) {
    int gn = tileN + s * 16 + l15;
    if (gn >= Nact) continue;
    float bv = (mode != 0 && bias) ? bias[gn] : 0.0f;
#pragma unroll
    for (int r = 0; r < 8; ++r) {
      int gm = tileM + wave * 16 + half * 8 + r;     // C/D layout per ISA §7.12.2
      if (gm >= M) continue;
      float v = acc[s][r] + bv;
      if (mode == 1) v = v > 0.f ? v : 0.f;
      Cmat[(long long)gm * ldc + gn] = v;
    }
  }
}

// ---------------- batch norm ----------------

__global__ void k_colstats(float* __restrict__ stats, const float* __restrict__ X,
                           int M, int Ncols) {
  int c = threadIdx.x;                 // blockDim.x == Ncols
  int r0 = blockIdx.x * 256;
  int r1 = r0 + 256; if (r1 > M) r1 = M;
  float s = 0.f, q = 0.f;
  for (int r = r0; r < r1; ++r) {
    float v = X[(long long)r * Ncols + c];
    s += v; q += v * v;
  }
  atomicAdd(&stats[c], s);
  atomicAdd(&stats[Ncols + c], q);
}

__global__ void k_bn_relu(float* __restrict__ Y, const float* __restrict__ X,
                          const float* __restrict__ stats,
                          const float* __restrict__ g, const float* __restrict__ b,
                          int M, int Ncols) {
  long long t = (long long)blockIdx.x * blockDim.x + threadIdx.x;
  if (t >= (long long)M * Ncols) return;
  int c = (int)(t % Ncols);
  float inv = 1.0f / (float)M;
  float m = stats[c] * inv;
  float var = stats[Ncols + c] * inv - m * m;
  float v = (X[t] - m) * rsqrtf(var + 1e-5f) * g[c] + b[c];
  Y[t] = v > 0.f ? v : 0.f;
}

// ---------------- pooling ----------------

__global__ void k_count(float* __restrict__ cnt, const int* __restrict__ batch, int n) {
  int i = blockIdx.x * blockDim.x + threadIdx.x;
  if (i < n) atomicAdd(&cnt[batch[i]], 1.0f);
}

__global__ void k_pool_div(float* __restrict__ pooled, const float* __restrict__ cnt,
                           int B, int C) {
  int t = blockIdx.x * blockDim.x + threadIdx.x;
  if (t >= B * C) return;
  float c = cnt[t / C];
  pooled[t] /= (c > 1.f ? c : 1.f);
}

// ---------------- host side ----------------

static inline long long cdiv(long long a, long long b) { return (a + b - 1) / b; }

struct Ctx {
  hipStream_t st;
  bf16_t* Abf;
  bf16_t* Wt;
  float*  stats;
};

static void run_gemm(const Ctx& c, float* Cmat, int ldc, const float* Asrc,
                     const float* W, int M, int N, int K,
                     const float* bias, int mode) {
  long long an = (long long)M * K;
  k_f32_to_bf16<<<(int)cdiv(an, 256), 256, 0, c.st>>>(c.Abf, Asrc, an);
  int Npad = (int)cdiv(N, 64) * 64;
  long long wn = (long long)Npad * K;
  k_wprep<<<(int)cdiv(wn, 256), 256, 0, c.st>>>(c.Wt, W, K, N, Npad);
  dim3 g((unsigned)cdiv(M, 128), (unsigned)(Npad / 64));
  size_t shmem = (size_t)64 * (K + 8) * sizeof(bf16_t);
  k_gemm_bf16<<<g, 256, shmem, c.st>>>(Cmat, ldc, c.Abf, c.Wt, M, N, K, bias, mode);
}

static void run_bn(const Ctx& c, float* Y, const float* X, int M, int Ncols,
                   const float* g, const float* b) {
  hipMemsetAsync(c.stats, 0, 2 * Ncols * sizeof(float), c.st);
  k_colstats<<<(int)cdiv(M, 256), Ncols, 0, c.st>>>(c.stats, X, M, Ncols);
  long long n = (long long)M * Ncols;
  k_bn_relu<<<(int)cdiv(n, 256), 256, 0, c.st>>>(Y, X, c.stats, g, b, M, Ncols);
}

struct MlpP { const float *W1, *g1, *b1, *W2, *g2, *b2; };

static void run_mlp2(const Ctx& c, float* out, const float* X, int M, int K,
                     const MlpP& p, float* Z, float* Hmid) {
  run_gemm(c, Z, 2 * kH, X, p.W1, M, 2 * kH, K, nullptr, 0);
  run_bn(c, Hmid, Z, M, 2 * kH, p.g1, p.b1);
  run_gemm(c, Z, kH, Hmid, p.W2, M, kH, 2 * kH, nullptr, 0);
  run_bn(c, out, Z, M, kH, p.g2, p.b2);
}

extern "C" void kernel_launch(void* const* d_in, const int* in_sizes, int n_in,
                              void* d_out, int out_size, void* d_ws, size_t ws_size,
                              hipStream_t stream) {
  (void)in_sizes; (void)n_in; (void)out_size; (void)ws_size;

  // ------- inputs (setup_inputs dict insertion order, recursive) -------
  int ci = 0;
  auto NF = [&]() { return (const float*)d_in[ci++]; };
  auto NI = [&]() { return (const int*)d_in[ci++]; };

  const float* emb_n = NF(); const float* emb_e = NF();
  const float* We1 = NF(); const float* be1 = NF();
  const float* We2 = NF(); const float* be2 = NF();
  const float* Wc1 = NF(); const float* bc1 = NF();
  const float* Wc2 = NF(); const float* bc2 = NF();

  struct LayerP { MlpP ee, ece, ecc; const float *mixW, *mixg, *mixb; } L[3];
  for (int l = 0; l < 3; ++l) {
    auto mlp = [&]() { MlpP m; m.W1=NF(); m.g1=NF(); m.b1=NF(); m.W2=NF(); m.g2=NF(); m.b2=NF(); return m; };
    L[l].ee = mlp(); L[l].ece = mlp(); L[l].ecc = mlp();
    L[l].mixW = NF(); L[l].mixg = NF(); L[l].mixb = NF();
  }
  const float* F1 = NF(); const float* Fb1 = NF(); const float* Fg1 = NF(); const float* Fbe1 = NF();
  const float* F2 = NF(); const float* Fb2 = NF(); const float* Fg2 = NF(); const float* Fbe2 = NF();
  const float* linW = NF(); const float* linb = NF();

  const int* node_x = NI(); const int* edge_x = NI();
  const int* e_vtx  = NI(); const int* c5_vtx = NI();
  const int* c6_vtx = NI(); const int* batch  = NI();

  // ------- workspace arena -------
  size_t off = 0;
  auto allocB = [&](size_t bytes) {
    void* p = (char*)d_ws + off;
    off = (off + bytes + 255) & ~(size_t)255;
    return p;
  };
  auto allocF = [&](long long n) { return (float*)allocB((size_t)n * 4); };

  float* node_rep = allocF((long long)kN * 64);
  float* edge_ft  = allocF((long long)kE * 64);
  float* edge_rep = allocF((long long)kE2 * 64);
  float* cyc      = allocF((long long)kR * 64);
  float* P_e = allocF((long long)kN * 64);
  float* Q_e = allocF((long long)kN * 64);
  float* D_e = allocF((long long)kE * 64);
  float* G1  = allocF((long long)kE2 * 128);
  float* P2  = allocF((long long)kN * 128);
  float* Q2  = allocF((long long)kN * 128);
  float* D2  = allocF((long long)kE * 128);
  float* CAT_EE = allocF((long long)kE2 * 320);
  float* EC5 = allocF((long long)kR5 * 128);
  float* EC6 = allocF((long long)kR6 * 128);
  float* P5  = allocF((long long)kN * 128);
  float* Q5  = allocF((long long)kN * 128);
  float* D56 = allocF((long long)kC5 * 128);
  float* CATC = allocF((long long)kR * 320);
  float* Pc  = allocF((long long)kN * 320);
  float* Qc  = allocF((long long)kN * 320);
  float* Dc  = allocF((long long)(2 * kC5) * 320);
  float* CAT_ECE = allocF((long long)kE2 * 704);
  float* eo_a = allocF((long long)kE2 * 64);
  float* eo_b = allocF((long long)kE2 * 64);
  float* MIXIN = allocF((long long)kE2 * 128);
  float* Zbuf = allocF((long long)kR * 128);
  float* Hmid = allocF((long long)kR * 128);
  float* reps = allocF((long long)kN * 128);
  float* pooled = allocF((long long)kB * 128);
  float* cnt  = allocF(kB);
  float* h1   = allocF((long long)kB * 256);
  float* h2   = allocF((long long)kB * 256);
  float* stats = allocF(640);
  bf16_t* Abf = (bf16_t*)allocB((size_t)kE2 * 704 * 2);
  bf16_t* Wt  = (bf16_t*)allocB((size_t)256 * 704 * 2);
  int* cyc_vtx = (int*)allocB((size_t)kR * 4);

  Ctx C{stream, Abf, Wt, stats};
  auto Z0 = [&](float* p, long long n) { hipMemsetAsync(p, 0, (size_t)n * 4, stream); };
  auto scat = [&](float* dst, int dld, int dcol, const int* darr, int dmode,
                  const float* src, int sld, int scol, const int* sarr, int smode,
                  int rows, int Cc) {
    long long tot = (long long)rows * Cc;
    k_gscatter_add<<<(int)cdiv(tot, 256), 256, 0, stream>>>(
        dst, dld, dcol, darr, dmode, src, sld, scol, sarr, smode, rows, Cc);
  };
  auto gath = [&](float* dst, int dld, int dcol, const float* src, int sld,
                  const int* idx, int mode, int rows, int Cc) {
    long long tot = (long long)rows * Cc;
    k_gather<<<(int)cdiv(tot, 256), 256, 0, stream>>>(
        dst, dld, dcol, src, sld, idx, mode, rows, Cc);
  };

  hipMemcpyAsync(cyc_vtx, c5_vtx, (size_t)kR5 * 4, hipMemcpyDeviceToDevice, stream);
  hipMemcpyAsync(cyc_vtx + kR5, c6_vtx, (size_t)kR6 * 4, hipMemcpyDeviceToDevice, stream);

  // ------- embeddings -------
  gath(node_rep, 64, 0, emb_n, 64, node_x, IM_ARR, kN, 64);
  gath(edge_ft,  64, 0, emb_e, 64, edge_x, IM_ARR, kE, 64);

  // ------- initial edge representation -------
  gath(G1, 128, 0, node_rep, 64, e_vtx, IM_ARR, kE2, 64);            // node2edge
  Z0(P_e, (long long)kN * 64);                                       // S
  scat(P_e, 64, 0, e_vtx, IM_ARR, edge_ft, 64, 0, nullptr, IM_HALF, kE2, 64);
  gath(G1, 128, 64, P_e, 64, e_vtx, IM_ARR, kE2, 64);                // edge2edge0
  run_gemm(C, Zbuf, 128, G1, We1, kE2, 128, 128, be1, 1);
  run_gemm(C, edge_rep, 64, Zbuf, We2, kE2, 64, 128, be2, 1);

  // ------- initial cycle representation: g11 on node features -------
  // c5 part
  Z0(P_e, (long long)kN * 64);
  scat(P_e, 64, 0, c5_vtx, IM_ARR, node_rep, 64, 0, c5_vtx, IM_ARR, kR5, 64);
  Z0(D_e, (long long)kC5 * 64);
  scat(D_e, 64, 0, nullptr, IM_FIFTH, node_rep, 64, 0, c5_vtx, IM_ARR, kR5, 64);
  Z0(Q_e, (long long)kN * 64);
  scat(Q_e, 64, 0, c5_vtx, IM_ARR, D_e, 64, 0, nullptr, IM_FIFTH, kR5, 64);
  gath(CATC, 128, 0,  P_e, 64, c5_vtx, IM_ARR, kR5, 64);             // CATC as [R,128]
  gath(CATC, 128, 64, Q_e, 64, c5_vtx, IM_ARR, kR5, 64);
  // c6 part
  Z0(P_e, (long long)kN * 64);
  scat(P_e, 64, 0, c6_vtx, IM_ARR, node_rep, 64, 0, c6_vtx, IM_ARR, kR6, 64);
  Z0(D_e, (long long)kC5 * 64);
  scat(D_e, 64, 0, nullptr, IM_SIXTH, node_rep, 64, 0, c6_vtx, IM_ARR, kR6, 64);
  Z0(Q_e, (long long)kN * 64);
  scat(Q_e, 64, 0, c6_vtx, IM_ARR, D_e, 64, 0, nullptr, IM_SIXTH, kR6, 64);
  gath(CATC + (long long)kR5 * 128, 128, 0,  P_e, 64, c6_vtx, IM_ARR, kR6, 64);
  gath(CATC + (long long)kR5 * 128, 128, 64, Q_e, 64, c6_vtx, IM_ARR, kR6, 64);
  run_gemm(C, Zbuf, 128, CATC, Wc1, kR, 128, 128, bc1, 1);
  run_gemm(C, cyc, 64, Zbuf, Wc2, kR, 64, 128, bc2, 1);

  // ------- layers -------
  for (int l = 0; l < 3; ++l) {
    // P/D/Q of edge_rep over (e_vtx, e_dom)  — shared by e1 and ec5/ec6
    Z0(P_e, (long long)kN * 64);
    scat(P_e, 64, 0, e_vtx, IM_ARR, edge_rep, 64, 0, nullptr, IM_IDENT, kE2, 64);
    Z0(D_e, (long long)kE * 64);
    scat(D_e, 64, 0, nullptr, IM_HALF, edge_rep, 64, 0, nullptr, IM_IDENT, kE2, 64);
    Z0(Q_e, (long long)kN * 64);
    scat(Q_e, 64, 0, e_vtx, IM_ARR, D_e, 64, 0, nullptr, IM_HALF, kE2, 64);

    // e1 = g11 -> [2E,128]
    gath(G1, 128, 0,  P_e, 64, e_vtx, IM_ARR, kE2, 64);
    gath(G1, 128, 64, Q_e, 64, e_vtx, IM_ARR, kE2, 64);
    // e1 = g11(e1) -> [2E,256], written straight into CAT_EE cols 64..319
    Z0(P2, (long long)kN * 128);
    scat(P2, 128, 0, e_vtx, IM_ARR, G1, 128, 0, nullptr, IM_IDENT, kE2, 128);
    Z0(D2, (long long)kE * 128);
    scat(D2, 128, 0, nullptr, IM_HALF, G1, 128, 0, nullptr, IM_IDENT, kE2, 128);
    Z0(Q2, (long long)kN * 128);
    scat(Q2, 128, 0, e_vtx, IM_ARR, D2, 128, 0, nullptr, IM_HALF, kE2, 128);
    gath(CAT_EE, 320, 0,   edge_rep, 64, nullptr, IM_IDENT, kE2, 64);
    gath(CAT_EE, 320, 64,  P2, 128, e_vtx, IM_ARR, kE2, 128);
    gath(CAT_EE, 320, 192, Q2, 128, e_vtx, IM_ARR, kE2, 128);
    run_mlp2(C, eo_a, CAT_EE, kE2, 320, L[l].ee, Zbuf, Hmid);

    // ec5/ec6 first g11 = gathers of shared P_e/Q_e at cycle vertices
    gath(EC5, 128, 0,  P_e, 64, c5_vtx, IM_ARR, kR5, 64);
    gath(EC5, 128, 64, Q_e, 64, c5_vtx, IM_ARR, kR5, 64);
    gath(EC6, 128, 0,  P_e, 64, c6_vtx, IM_ARR, kR6, 64);
    gath(EC6, 128, 64, Q_e, 64, c6_vtx, IM_ARR, kR6, 64);
    // second g11 over 5-cycles -> CATC rows [0,R5) cols 0..255
    Z0(P5, (long long)kN * 128);
    scat(P5, 128, 0, c5_vtx, IM_ARR, EC5, 128, 0, nullptr, IM_IDENT, kR5, 128);
    Z0(D56, (long long)kC5 * 128);
    scat(D56, 128, 0, nullptr, IM_FIFTH, EC5, 128, 0, nullptr, IM_IDENT, kR5, 128);
    Z0(Q5, (long long)kN * 128);
    scat(Q5, 128, 0, c5_vtx, IM_ARR, D56, 128, 0, nullptr, IM_FIFTH, kR5, 128);
    gath(CATC, 320, 0,   P5, 128, c5_vtx, IM_ARR, kR5, 128);
    gath(CATC, 320, 128, Q5, 128, c5_vtx, IM_ARR, kR5, 128);
    // second g11 over 6-cycles -> CATC rows [R5,R) cols 0..255 (reuse P5/Q5/D56)
    Z0(P5, (long long)kN * 128);
    scat(P5, 128, 0, c6_vtx, IM_ARR, EC6, 128, 0, nullptr, IM_IDENT, kR6, 128);
    Z0(D56, (long long)kC5 * 128);
    scat(D56, 128, 0, nullptr, IM_SIXTH, EC6, 128, 0, nullptr, IM_IDENT, kR6, 128);
    Z0(Q5, (long long)kN * 128);
    scat(Q5, 128, 0, c6_vtx, IM_ARR, D56, 128, 0, nullptr, IM_SIXTH, kR6, 128);
    gath(CATC + (long long)kR5 * 320, 320, 0,   P5, 128, c6_vtx, IM_ARR, kR6, 128);
    gath(CATC + (long long)kR5 * 320, 320, 128, Q5, 128, c6_vtx, IM_ARR, kR6, 128);
    // cycle_new cols 256..319 = old cyc
    gath(CATC, 320, 256, cyc, 64, nullptr, IM_IDENT, kR, 64);

    // c2e = g11(cycle_new, cyc_vtx, cyc_dom, C5+C6, e_vtx)
    Z0(Pc, (long long)kN * 320);
    scat(Pc, 320, 0, cyc_vtx, IM_ARR, CATC, 320, 0, nullptr, IM_IDENT, kR, 320);
    Z0(Dc, (long long)(2 * kC5) * 320);
    scat(Dc, 320, 0, nullptr, IM_CYC, CATC, 320, 0, nullptr, IM_IDENT, kR, 320);
    Z0(Qc, (long long)kN * 320);
    scat(Qc, 320, 0, cyc_vtx, IM_ARR, Dc, 320, 0, nullptr, IM_CYC, kR, 320);
    gath(CAT_ECE, 704, 0,   edge_rep, 64, nullptr, IM_IDENT, kE2, 64);
    gath(CAT_ECE, 704, 64,  Pc, 320, e_vtx, IM_ARR, kE2, 320);
    gath(CAT_ECE, 704, 384, Qc, 320, e_vtx, IM_ARR, kE2, 320);
    run_mlp2(C, eo_b, CAT_ECE, kE2, 704, L[l].ece, Zbuf, Hmid);

    // cyc update
    run_mlp2(C, cyc, CATC, kR, 320, L[l].ecc, Zbuf, Hmid);

    // mix
    gath(MIXIN, 128, 0,  eo_a, 64, nullptr, IM_IDENT, kE2, 64);
    gath(MIXIN, 128, 64, eo_b, 64, nullptr, IM_IDENT, kE2, 64);
    run_gemm(C, Zbuf, 64, MIXIN, L[l].mixW, kE2, 64, 128, nullptr, 0);
    run_bn(C, edge_rep, Zbuf, kE2, 64, L[l].mixg, L[l].mixb);
  }

  // ------- readout + pooling -------
  Z0(reps, (long long)kN * 128);
  scat(reps, 128, 0,  e_vtx,   IM_ARR, edge_rep, 64, 0, nullptr, IM_IDENT, kE2, 64);
  scat(reps, 128, 64, cyc_vtx, IM_ARR, cyc,      64, 0, nullptr, IM_IDENT, kR,  64);
  Z0(pooled, (long long)kB * 128);
  Z0(cnt, kB);
  k_count<<<(int)cdiv(kN, 256), 256, 0, stream>>>(cnt, batch, kN);
  scat(pooled, 128, 0, batch, IM_ARR, reps, 128, 0, nullptr, IM_IDENT, kN, 128);
  k_pool_div<<<(int)cdiv(kB * 128, 256), 256, 0, stream>>>(pooled, cnt, kB, 128);

  // ------- head MLP -------
  run_gemm(C, Zbuf, 256, pooled, F1, kB, 256, 128, Fb1, 2);
  run_bn(C, h1, Zbuf, kB, 256, Fg1, Fbe1);
  run_gemm(C, Zbuf, 256, h1, F2, kB, 256, 256, Fb2, 2);
  run_bn(C, h2, Zbuf, kB, 256, Fg2, Fbe2);
  run_gemm(C, (float*)d_out, kOut, h2, linW, kB, kOut, 256, linb, 2);
}
